// SelfAttention_13623636263698
// MI455X (gfx1250) — matrix operand order repaired and checked by reading, not compiled, not run on hardware
//
#include <hip/hip_runtime.h>
#include <hip/hip_bf16.h>
#include <math.h>

// Shapes (compile-time constants from the reference)
constexpr int Bc = 4, Sc = 2048, Dc = 1024, Hc = 16, HDc = 64;

typedef __attribute__((ext_vector_type(16))) _Float16 v16h;
typedef __attribute__((ext_vector_type(8)))  _Float16 v8h;
typedef __attribute__((ext_vector_type(8)))  float    v8f;

// Fold softmax scale (1/sqrt(64)) and log2(e) into Q so softmax uses exp2.
#define QSCALE (0.125f * 1.44269504088896340736f)

static __device__ __forceinline__ v8f wmma_f16(v16h a, v16h b, v8f c) {
  return __builtin_amdgcn_wmma_f32_16x16x32_f16(
      /*neg_a=*/false, a, /*neg_b=*/false, b,
      /*c_mod=*/(short)0, c, /*reuse_a=*/false, /*reuse_b=*/false);
}

// ---------------------------------------------------------------------------
// Kernel 1: fused QKV projection.  grid = (N/64, M/128, 3), block = 128.
// 4 waves; each wave computes a 32x64 strip (2 M-tiles x 4 N-tiles = 8 WMMAs
// per K-step) of the 128x64 block tile.  blockIdx.z selects {Q,K,V}.
// Outputs stored as f16, head-major:
//   Q: [b*H+h][s][hd]  (pre-scaled by QSCALE)
//   K: [b*H+h][hd][s]  (transposed -> contiguous B-fragments for Q@K^T)
//   V: [b*H+h][s][hd]
// ---------------------------------------------------------------------------
__global__ __launch_bounds__(128) void qkv_proj_kernel(
    const float* __restrict__ x,
    const float* __restrict__ Wq, const float* __restrict__ bq,
    const float* __restrict__ Wk, const float* __restrict__ bk,
    const float* __restrict__ Wv, const float* __restrict__ bv,
    _Float16* __restrict__ qws, _Float16* __restrict__ kws,
    _Float16* __restrict__ vws)
{
  const int z = blockIdx.z;
  const float* __restrict__ W  = (z == 0) ? Wq : (z == 1) ? Wk : Wv;
  const float* __restrict__ bs = (z == 0) ? bq : (z == 1) ? bk : bv;

  const int n0    = blockIdx.x * 64;
  const int m0    = blockIdx.y * 128;
  const int tid   = threadIdx.x;
  const int wave  = tid >> 5;
  const int lane  = tid & 31;
  const int laneN = lane & 15;       // C-fragment N index
  const int rA    = lane & 15;       // A-fragment row within tile
  const int kb    = (lane >> 4) * 8; // A-fragment K sub-base / C-fragment M base

  __shared__ __align__(32) _Float16 wtile[32 * 64];  // W[k0..k0+32)[n0..n0+64)

  v8f acc[2][4] = {{v8f{}, v8f{}, v8f{}, v8f{}}, {v8f{}, v8f{}, v8f{}, v8f{}}};

  for (int k0 = 0; k0 < Dc; k0 += 32) {
    // --- cooperatively stage the 32x64 weight tile into LDS (fp32 -> f16) ---
    {
      const int i = tid * 16;  // 128 threads x 16 elements = 2048
      const int r = i >> 6;
      const int c = i & 63;
      const float* __restrict__ wp = W + (size_t)(k0 + r) * Dc + n0 + c;
      v16h t;
#pragma unroll
      for (int u = 0; u < 16; ++u) t[u] = (_Float16)wp[u];
      *(v16h*)&wtile[r * 64 + c] = t;
    }
    __syncthreads();

    // --- A fragments: two 16x32 strips per wave (ISA 16-bit A layout) ------
    v16h a[2];
#pragma unroll
    for (int ms = 0; ms < 2; ++ms) {
      const float* __restrict__ xp =
          x + (size_t)(m0 + wave * 32 + ms * 16 + rA) * Dc + k0 + kb;
#pragma unroll
      for (int u = 0; u < 8; ++u) a[ms][u] = (_Float16)xp[u];
#pragma unroll
      for (int u = 0; u < 8; ++u) a[ms][8 + u] = (_Float16)xp[16 + u];
    }

    // --- B fragments from LDS: B[k=table order][n=lane&15] ------
#pragma unroll
    for (int nt = 0; nt < 4; ++nt) {
      v16h bfr;
#pragma unroll
      for (int u = 0; u < 8; ++u) {
        bfr[u]     = wtile[(kb + u) * 64 + nt * 16 + laneN];
        bfr[8 + u] = wtile[(16 + kb + u) * 64 + nt * 16 + laneN];
      }
      acc[0][nt] = wmma_f16(a[0], bfr, acc[0][nt]);
      acc[1][nt] = wmma_f16(a[1], bfr, acc[1][nt]);
    }
    __syncthreads();
  }

  // --- epilogue: bias + convert + scatter.  One z-branch, no inner branches.
  if (z == 1) {
    // K path: [hd][s] layout -> 8 consecutive f16 per (nt,ms) -> b128 store
#pragma unroll
    for (int ms = 0; ms < 2; ++ms) {
      const int mrow = m0 + wave * 32 + ms * 16 + kb;  // first of 8 rows
      const int b    = mrow >> 11;
      const int s    = mrow & 2047;
#pragma unroll
      for (int nt = 0; nt < 4; ++nt) {
        const int col  = n0 + nt * 16 + laneN;
        const float bb = bs[col];
        const int h  = col >> 6;
        const int hd = col & 63;
        v8h pack;
#pragma unroll
        for (int v = 0; v < 8; ++v) pack[v] = (_Float16)(acc[ms][nt][v] + bb);
        *(v8h*)&kws[((size_t)(b * Hc + h)) * Sc * HDc + (size_t)hd * Sc + s] =
            pack;
      }
    }
  } else {
    _Float16* __restrict__ dst = (z == 0) ? qws : vws;
    const float scale = (z == 0) ? QSCALE : 1.0f;
#pragma unroll
    for (int ms = 0; ms < 2; ++ms) {
#pragma unroll
      for (int nt = 0; nt < 4; ++nt) {
        const int col  = n0 + nt * 16 + laneN;
        const float bb = bs[col];
        const int h  = col >> 6;
        const int hd = col & 63;
#pragma unroll
        for (int v = 0; v < 8; ++v) {
          const int mrow = m0 + wave * 32 + ms * 16 + kb + v;
          const int b    = mrow >> 11;
          const int s    = mrow & 2047;
          dst[((size_t)(b * Hc + h)) * Sc * HDc + (size_t)s * HDc + hd] =
              (_Float16)((acc[ms][nt][v] + bb) * scale);
        }
      }
    }
  }
}

// ---------------------------------------------------------------------------
// Kernel 2: flash attention.  grid = (S/64, B*H), block = 128 (4 waves).
// Each wave owns 16 query rows.  K and V 64x64 tiles are staged once per
// key-tile into LDS (shared by all 4 waves -> 4x less global traffic);
// P is restriped C-layout -> A-layout through per-wave LDS.
// ---------------------------------------------------------------------------
__global__ __launch_bounds__(128) void attn_kernel(
    const _Float16* __restrict__ qws, const _Float16* __restrict__ kws,
    const _Float16* __restrict__ vws, float* __restrict__ out)
{
  const int bh   = blockIdx.y;  // 0..63
  const int b    = bh >> 4;
  const int h    = bh & 15;
  const int tid  = threadIdx.x;
  const int wave = tid >> 5;
  const int lane = tid & 31;
  const int rA   = lane & 15;
  const int kb   = (lane >> 4) * 8;
  const int s0   = blockIdx.x * 64 + wave * 16;  // first query row of wave
  const size_t base = (size_t)bh * Sc * HDc;

  // LDS: K^T tile [hd][key], V tile [key][hd]; stride 72 f16 = 144 B
  // (16B-aligned for ds_load_b128, 36-bank stagger).  P: per-wave 16x64.
  __shared__ __align__(16) _Float16 kt_lds[64][72];
  __shared__ __align__(16) _Float16 v_lds[64][72];
  __shared__ _Float16 plds[4][16][68];

  // Cooperative staging role: thread -> (row, 32-col half)
  const int sr  = tid >> 1;
  const int sc2 = (tid & 1) * 32;

  // --- Q fragments (A-layout), HD=64 split into two k=32 fragments ---------
  v16h qf[2];
#pragma unroll
  for (int j = 0; j < 2; ++j) {
    const _Float16* qp = qws + base + (size_t)(s0 + rA) * HDc + j * 32 + kb;
    v8h lo = *(const v8h*)qp;
    v8h hi = *(const v8h*)(qp + 16);
#pragma unroll
    for (int u = 0; u < 8; ++u) { qf[j][u] = lo[u]; qf[j][8 + u] = hi[u]; }
  }

  float mrow[8], lrow[8];
  v8f   oacc[4] = {v8f{}, v8f{}, v8f{}, v8f{}};
#pragma unroll
  for (int v = 0; v < 8; ++v) { mrow[v] = -1e30f; lrow[v] = 0.0f; }

  for (int kt = 0; kt < Sc; kt += 64) {
    // ---------------- stage K^T and V tiles into LDS -----------------------
    __syncthreads();  // previous iteration's LDS consumers are done
    {
      const _Float16* kp = kws + base + (size_t)sr * Sc + kt + sc2;
      *(v16h*)&kt_lds[sr][sc2]      = *(const v16h*)kp;
      *(v16h*)&kt_lds[sr][sc2 + 16] = *(const v16h*)(kp + 16);
      const _Float16* vp = vws + base + (size_t)(kt + sr) * HDc + sc2;
      *(v16h*)&v_lds[sr][sc2]       = *(const v16h*)vp;
      *(v16h*)&v_lds[sr][sc2 + 16]  = *(const v16h*)(vp + 16);
      if (kt + 64 < Sc) {  // prefetch next key tile (global_prefetch_b8)
        __builtin_prefetch(kws + base + (size_t)sr * Sc + kt + 64 + sc2, 0, 1);
        __builtin_prefetch(vws + base + (size_t)(kt + 64 + sr) * HDc + sc2, 0, 1);
      }
    }
    __syncthreads();

    // ---------------- S = Q @ K^T  (scores, log2 domain) -------------------
    v8f sc[4] = {v8f{}, v8f{}, v8f{}, v8f{}};
#pragma unroll
    for (int j = 0; j < 2; ++j) {
#pragma unroll
      for (int nt = 0; nt < 4; ++nt) {
        v16h bfr;
#pragma unroll
        for (int u = 0; u < 8; ++u) {
          bfr[u]     = kt_lds[j * 32 + kb + u][nt * 16 + rA];
          bfr[8 + u] = kt_lds[j * 32 + 16 + kb + u][nt * 16 + rA];
        }
        sc[nt] = wmma_f16(qf[j], bfr, sc[nt]);
      }
    }

    // ---------------- online softmax (per C-layout row) --------------------
    float pr[4][8];
#pragma unroll
    for (int v = 0; v < 8; ++v) {
      float mx = fmaxf(fmaxf(sc[0][v], sc[1][v]), fmaxf(sc[2][v], sc[3][v]));
#pragma unroll
      for (int off = 8; off >= 1; off >>= 1)
        mx = fmaxf(mx, __shfl_xor(mx, off, 32));  // reduce 16-lane half
      const float mn   = fmaxf(mrow[v], mx);
      const float corr = exp2f(mrow[v] - mn);
      mrow[v] = mn;
      float rs = 0.0f;
#pragma unroll
      for (int nt = 0; nt < 4; ++nt) {
        const float p = exp2f(sc[nt][v] - mn);
        pr[nt][v] = p;
        rs += p;
      }
#pragma unroll
      for (int off = 8; off >= 1; off >>= 1)
        rs += __shfl_xor(rs, off, 32);
      lrow[v] = lrow[v] * corr + rs;
#pragma unroll
      for (int nt = 0; nt < 4; ++nt) oacc[nt][v] *= corr;
    }

    // ------------- restripe P: C/D layout -> A layout via LDS --------------
#pragma unroll
    for (int nt = 0; nt < 4; ++nt)
#pragma unroll
      for (int v = 0; v < 8; ++v)
        plds[wave][kb + v][nt * 16 + (lane & 15)] = (_Float16)pr[nt][v];
    __syncthreads();

    // ---------------- O += P @ V -------------------------------------------
#pragma unroll
    for (int j = 0; j < 2; ++j) {
      v16h pf;
#pragma unroll
      for (int u = 0; u < 8; ++u) {
        pf[u]     = plds[wave][rA][j * 32 + kb + u];
        pf[8 + u] = plds[wave][rA][j * 32 + kb + 16 + u];
      }
#pragma unroll
      for (int nt = 0; nt < 4; ++nt) {
        v16h bfr;
#pragma unroll
        for (int u = 0; u < 8; ++u) {
          bfr[u]     = v_lds[j * 32 + kb + u][nt * 16 + rA];
          bfr[8 + u] = v_lds[j * 32 + 16 + kb + u][nt * 16 + rA];
        }
        oacc[nt] = wmma_f16(pf, bfr, oacc[nt]);
      }
    }
  }

  // ---------------- normalize and store [B,S,D] fp32 -----------------------
#pragma unroll
  for (int nt = 0; nt < 4; ++nt) {
#pragma unroll
    for (int v = 0; v < 8; ++v) {
      const int s = s0 + kb + v;
      out[((size_t)(b * Sc + s)) * Dc + h * 64 + nt * 16 + (lane & 15)] =
          oacc[nt][v] / lrow[v];
    }
  }
}

// ---------------------------------------------------------------------------
extern "C" void kernel_launch(void* const* d_in, const int* in_sizes, int n_in,
                              void* d_out, int out_size, void* d_ws, size_t ws_size,
                              hipStream_t stream) {
  const float* x  = (const float*)d_in[0];
  const float* Wq = (const float*)d_in[1];
  const float* bq = (const float*)d_in[2];
  const float* Wk = (const float*)d_in[3];
  const float* bk = (const float*)d_in[4];
  const float* Wv = (const float*)d_in[5];
  const float* bv = (const float*)d_in[6];

  const size_t elems = (size_t)Bc * Hc * Sc * HDc;  // 8,388,608 per tensor
  _Float16* qws = (_Float16*)d_ws;
  _Float16* kws = qws + elems;
  _Float16* vws = kws + elems;

  // Projections: M = B*S = 8192 rows, N = 1024 cols, block tile 128x64
  dim3 g1(Dc / 64, (Bc * Sc) / 128, 3);
  qkv_proj_kernel<<<g1, 128, 0, stream>>>(x, Wq, bq, Wk, bk, Wv, bv,
                                          qws, kws, vws);

  // Attention: 32 query tiles x 64 (b,h) pairs
  dim3 g2(Sc / 64, Bc * Hc);
  attn_kernel<<<g2, 128, 0, stream>>>(qws, kws, vws, (float*)d_out);
}
